// RegionAttention_69836168233139
// MI455X (gfx1250) — matrix operand order repaired and checked by reading, not compiled, not run on hardware
//
#include <hip/hip_runtime.h>
#include <hip/hip_bf16.h>

typedef __bf16 bf16;
typedef bf16 v16bf __attribute__((ext_vector_type(16)));
typedef bf16 v8bf  __attribute__((ext_vector_type(8)));
typedef float v8f  __attribute__((ext_vector_type(8)));
typedef float v4f  __attribute__((ext_vector_type(4)));

#define LOG2E 1.4426950408889634f

// workspace layout in bf16 elements (total ~75.5 MB)
static constexpr size_t XN_OFF = 0;                      // 16384 x 256
static constexpr size_t Q_OFF  = 4194304;                // 128 probs x 1024 x 64
static constexpr size_t K_OFF  = Q_OFF + 8388608;
static constexpr size_t V_OFF  = K_OFF + 8388608;
static constexpr size_t O_OFF  = V_OFF + 8388608;        // 16384 x 512

__device__ __forceinline__ v8f wmma_bf16(v16bf a, v16bf b, v8f c) {
  return __builtin_amdgcn_wmma_f32_16x16x32_bf16(false, a, false, b, (short)0, c,
                                                 false, false);
}

// load 16 bf16 as two 8-element (16B) chunks -> one WMMA fragment register set
__device__ __forceinline__ v16bf ld_frag(const bf16* p_lo, const bf16* p_hi) {
  v8bf lo = *(const v8bf*)p_lo;
  v8bf hi = *(const v8bf*)p_hi;
  v16bf f;
#pragma unroll
  for (int i = 0; i < 8; ++i) { f[i] = lo[i]; f[i + 8] = hi[i]; }
  return f;
}

// ---------------- Kernel 1: LayerNorm -> bf16 xn ----------------
__global__ __launch_bounds__(256) void ln_k(const float* __restrict__ x,
                                            const float* __restrict__ lw,
                                            const float* __restrict__ lb,
                                            bf16* __restrict__ xn) {
  int lane = threadIdx.x & 31, wv = threadIdx.x >> 5;
  int c0 = lane * 8;
  v4f w0 = *(const v4f*)(lw + c0), w1 = *(const v4f*)(lw + c0 + 4);
  v4f b0 = *(const v4f*)(lb + c0), b1 = *(const v4f*)(lb + c0 + 4);
#pragma unroll
  for (int it = 0; it < 4; ++it) {
    int t = blockIdx.x * 32 + wv * 4 + it;
    const float* row = x + (size_t)t * 256 + c0;
    v4f a0 = *(const v4f*)(row);
    v4f a1 = *(const v4f*)(row + 4);
    float s = 0.f, s2 = 0.f;
#pragma unroll
    for (int i = 0; i < 4; ++i) {
      s += a0[i] + a1[i];
      s2 += a0[i] * a0[i] + a1[i] * a1[i];
    }
#pragma unroll
    for (int m = 16; m >= 1; m >>= 1) {
      s += __shfl_xor(s, m, 32);
      s2 += __shfl_xor(s2, m, 32);
    }
    float mu = s * (1.f / 256.f);
    float var = s2 * (1.f / 256.f) - mu * mu;
    float rs = rsqrtf(var + 1e-5f);
    v8bf o;
#pragma unroll
    for (int i = 0; i < 4; ++i) {
      o[i]     = (bf16)((a0[i] - mu) * rs * w0[i] + b0[i]);
      o[i + 4] = (bf16)((a1[i] - mu) * rs * w1[i] + b1[i]);
    }
    *(v8bf*)(xn + (size_t)t * 256 + c0) = o;
  }
}

// ---------------- Kernel 2: QKV GEMM (xn @ Wqkv^T) -> Q/K/V layouts ----------------
// Q part is pre-scaled by softmax scale (1/8), so attention skips the mul.
__global__ __launch_bounds__(256) void qkv_k(const bf16* __restrict__ xn,
                                             const float* __restrict__ wq,
                                             bf16* __restrict__ ws) {
  __shared__ bf16 As[128][136];   // token x k-chunk
  __shared__ bf16 Bs[64][136];    // feature x k-chunk
  int tid = threadIdx.x, lane = tid & 31, wv = tid >> 5;
  int l16 = lane & 15, hi = lane >> 4, h8 = hi * 8;
  int mt = blockIdx.x, nt = blockIdx.y;

  v8bf ar[8];
  float wr[32];
  auto fetch = [&](int kc) {
#pragma unroll
    for (int i = 0; i < 8; ++i) {
      int idx = tid + 256 * i;
      int rr = idx >> 4, cc = (idx & 15) * 8;
      ar[i] = *(const v8bf*)(xn + (size_t)(mt * 128 + rr) * 256 + kc * 128 + cc);
    }
#pragma unroll
    for (int i = 0; i < 32; ++i) {
      int idx = tid + 256 * i;
      int rr = idx >> 7, cc = idx & 127;
      wr[i] = wq[(size_t)(nt * 64 + rr) * 256 + kc * 128 + cc];
    }
  };
  auto commit = [&]() {
#pragma unroll
    for (int i = 0; i < 8; ++i) {
      int idx = tid + 256 * i;
      *(v8bf*)&As[idx >> 4][(idx & 15) * 8] = ar[i];
    }
#pragma unroll
    for (int i = 0; i < 32; ++i) {
      int idx = tid + 256 * i;
      Bs[idx >> 7][idx & 127] = (bf16)wr[i];
    }
  };

  v8f acc[4] = {};
  fetch(0);
  for (int kc = 0; kc < 2; ++kc) {
    __syncthreads();
    commit();
    __syncthreads();
    if (kc < 1) fetch(kc + 1);   // overlap next-chunk global loads with WMMAs
#pragma unroll
    for (int ks = 0; ks < 4; ++ks) {
      const bf16* arow = &As[wv * 16 + l16][ks * 32];
      v16bf af = ld_frag(arow + 2 * h8, arow + 2 * h8 + 8);
#pragma unroll
      for (int nn = 0; nn < 4; ++nn) {
        const bf16* brow = &Bs[nn * 16 + l16][ks * 32 + hi * 16];
        acc[nn] = wmma_bf16(af, ld_frag(brow, brow + 8), acc[nn]);
      }
    }
  }
  // scatter into per-(region,head) [1024 x 64] bf16
  int p = nt >> 3, head = nt & 7;
  int breg = mt >> 3;                       // 128 token rows lie in one region
  int ibase = (mt & 7) * 128 + wv * 16;
  float sc = (p == 0) ? 0.125f : 1.0f;      // fold attention scale into Q
  bf16* dst = ws + Q_OFF + (size_t)p * 8388608 +
              (size_t)(breg * 8 + head) * 65536;
#pragma unroll
  for (int nn = 0; nn < 4; ++nn) {
    int dcol = nn * 16 + l16;
#pragma unroll
    for (int r = 0; r < 8; ++r) {
      int i = ibase + r + h8;
      dst[(size_t)i * 64 + dcol] = (bf16)(acc[nn][r] * sc);
    }
  }
}

// ---------------- Kernel 3: flash attention per (problem, q-block) ----------------
__global__ __launch_bounds__(256) void attn_k(const bf16* __restrict__ wsc,
                                              bf16* __restrict__ ws) {
  __shared__ bf16 Ks[64][72];       // kv-row major
  __shared__ bf16 Vt[64][72];       // transposed: [d][kv]
  __shared__ bf16 Pw[8][16][72];    // wave-private P staging
  int tid = threadIdx.x, lane = tid & 31, wv = tid >> 5;
  int l16 = lane & 15, hi = lane >> 4, h8 = hi * 8;
  int qb = blockIdx.x, prob = blockIdx.y;
  const bf16* Q  = wsc + Q_OFF + (size_t)prob * 65536;
  const bf16* Kg = wsc + K_OFF + (size_t)prob * 65536;
  const bf16* Vg = wsc + V_OFF + (size_t)prob * 65536;

  const bf16* qrow = Q + (size_t)(qb * 128 + wv * 16 + l16) * 64;
  v16bf qf[2];
#pragma unroll
  for (int kd = 0; kd < 2; ++kd)
    qf[kd] = ld_frag(qrow + kd * 32 + 2 * h8, qrow + kd * 32 + 2 * h8 + 8);

  // register double-buffer for K/V global->LDS staging
  v8bf kr[2], vr[2];
  auto fetch = [&](int j) {
#pragma unroll
    for (int i = 0; i < 2; ++i) {
      int idx = tid + 256 * i;
      kr[i] = *(const v8bf*)(Kg + (size_t)(j * 64 + (idx >> 3)) * 64 + (idx & 7) * 8);
      vr[i] = *(const v8bf*)(Vg + (size_t)(j * 64 + (idx & 63)) * 64 + (idx >> 6) * 8);
    }
  };
  auto commit = [&]() {
#pragma unroll
    for (int i = 0; i < 2; ++i) {
      int idx = tid + 256 * i;
      *(v8bf*)&Ks[idx >> 3][(idx & 7) * 8] = kr[i];
      int rr = idx & 63, c8 = (idx >> 6) * 8;  // conflict-free transpose scatter:
#pragma unroll                                 // per store: one row, 32 contiguous cols
      for (int e = 0; e < 8; ++e) Vt[c8 + e][rr] = vr[i][e];
    }
  };

  float m[8], l[8];
  v8f oacc[4] = {};
#pragma unroll
  for (int r = 0; r < 8; ++r) { m[r] = -3.0e38f; l[r] = 0.f; }

  fetch(0);
  for (int j = 0; j < 16; ++j) {
    __syncthreads();
    commit();
    __syncthreads();
    if (j < 15) fetch(j + 1);    // overlap next block's global loads with compute

    // S = Q K^T  (scale pre-folded into Q)
    v8f s[4] = {};
#pragma unroll
    for (int nn = 0; nn < 4; ++nn)
#pragma unroll
      for (int kd = 0; kd < 2; ++kd) {
        const bf16* brow = &Ks[nn * 16 + l16][kd * 32 + hi * 16];
        s[nn] = wmma_bf16(qf[kd], ld_frag(brow, brow + 8), s[nn]);
      }
    float mcur[8];
#pragma unroll
    for (int r = 0; r < 8; ++r) mcur[r] = -3.0e38f;
#pragma unroll
    for (int nn = 0; nn < 4; ++nn)
#pragma unroll
      for (int r = 0; r < 8; ++r) mcur[r] = fmaxf(mcur[r], s[nn][r]);
#pragma unroll
    for (int r = 0; r < 8; ++r)
#pragma unroll
      for (int mk = 1; mk < 16; mk <<= 1)
        mcur[r] = fmaxf(mcur[r], __shfl_xor(mcur[r], mk, 16));
    float alpha[8], rsum[8];
#pragma unroll
    for (int r = 0; r < 8; ++r) {
      float mn = fmaxf(m[r], mcur[r]);
      alpha[r] = exp2f((m[r] - mn) * LOG2E);
      m[r] = mn;
      rsum[r] = 0.f;
    }
#pragma unroll
    for (int nn = 0; nn < 4; ++nn)
#pragma unroll
      for (int r = 0; r < 8; ++r) {
        float p = exp2f((s[nn][r] - m[r]) * LOG2E);
        rsum[r] += p;
        Pw[wv][r + h8][nn * 16 + l16] = (bf16)p;
      }
#pragma unroll
    for (int r = 0; r < 8; ++r) {
#pragma unroll
      for (int mk = 1; mk < 16; mk <<= 1) rsum[r] += __shfl_xor(rsum[r], mk, 16);
      l[r] = l[r] * alpha[r] + rsum[r];
    }
#pragma unroll
    for (int dn = 0; dn < 4; ++dn)
#pragma unroll
      for (int r = 0; r < 8; ++r) oacc[dn][r] *= alpha[r];

    // O += P V  (wave-local LDS round trip for A-fragment layout)
    v16bf pf[2];
#pragma unroll
    for (int kp = 0; kp < 2; ++kp) {
      const bf16* prow = &Pw[wv][l16][kp * 32];
      pf[kp] = ld_frag(prow + 2 * h8, prow + 2 * h8 + 8);
    }
#pragma unroll
    for (int dn = 0; dn < 4; ++dn)
#pragma unroll
      for (int kp = 0; kp < 2; ++kp) {
        const bf16* vrow = &Vt[dn * 16 + l16][kp * 32 + hi * 16];
        oacc[dn] = wmma_bf16(pf[kp], ld_frag(vrow, vrow + 8), oacc[dn]);
      }
  }

  int breg = prob >> 3, head = prob & 7;
  bf16* Ob = ws + O_OFF;
#pragma unroll
  for (int r = 0; r < 8; ++r) {
    float inv = 1.f / l[r];
    int t = breg * 1024 + qb * 128 + wv * 16 + r + h8;
#pragma unroll
    for (int dn = 0; dn < 4; ++dn)
      Ob[(size_t)t * 512 + head * 64 + dn * 16 + l16] = (bf16)(oacc[dn][r] * inv);
  }
}

// ---------------- Kernel 4: out projection (O @ Wout^T + b) ----------------
__global__ __launch_bounds__(256) void out_k(const bf16* __restrict__ ws,
                                             const float* __restrict__ wo,
                                             const float* __restrict__ bo,
                                             float* __restrict__ y) {
  __shared__ bf16 As[128][136];
  __shared__ bf16 Bs[64][136];
  int tid = threadIdx.x, lane = tid & 31, wv = tid >> 5;
  int l16 = lane & 15, hi = lane >> 4, h8 = hi * 8;
  int mt = blockIdx.x, nt = blockIdx.y;
  const bf16* O = ws + O_OFF;

  v8bf ar[8];
  float wr[32];
  auto fetch = [&](int kc) {
#pragma unroll
    for (int i = 0; i < 8; ++i) {
      int idx = tid + 256 * i;
      int rr = idx >> 4, cc = (idx & 15) * 8;
      ar[i] = *(const v8bf*)(O + (size_t)(mt * 128 + rr) * 512 + kc * 128 + cc);
    }
#pragma unroll
    for (int i = 0; i < 32; ++i) {
      int idx = tid + 256 * i;
      int rr = idx >> 7, cc = idx & 127;
      wr[i] = wo[(size_t)(nt * 64 + rr) * 512 + kc * 128 + cc];
    }
  };
  auto commit = [&]() {
#pragma unroll
    for (int i = 0; i < 8; ++i) {
      int idx = tid + 256 * i;
      *(v8bf*)&As[idx >> 4][(idx & 15) * 8] = ar[i];
    }
#pragma unroll
    for (int i = 0; i < 32; ++i) {
      int idx = tid + 256 * i;
      Bs[idx >> 7][idx & 127] = (bf16)wr[i];
    }
  };

  v8f acc[4] = {};
  fetch(0);
  for (int kc = 0; kc < 4; ++kc) {
    __syncthreads();
    commit();
    __syncthreads();
    if (kc < 3) fetch(kc + 1);
#pragma unroll
    for (int ks = 0; ks < 4; ++ks) {
      const bf16* arow = &As[wv * 16 + l16][ks * 32];
      v16bf af = ld_frag(arow + 2 * h8, arow + 2 * h8 + 8);
#pragma unroll
      for (int nn = 0; nn < 4; ++nn) {
        const bf16* brow = &Bs[nn * 16 + l16][ks * 32 + hi * 16];
        acc[nn] = wmma_bf16(af, ld_frag(brow, brow + 8), acc[nn]);
      }
    }
  }
#pragma unroll
  for (int nn = 0; nn < 4; ++nn) {
    int e = nt * 64 + nn * 16 + l16;
    float bias = bo[e];
#pragma unroll
    for (int r = 0; r < 8; ++r) {
      int t = mt * 128 + wv * 16 + r + h8;
      y[(size_t)t * 256 + e] = acc[nn][r] + bias;
    }
  }
}

extern "C" void kernel_launch(void* const* d_in, const int* in_sizes, int n_in,
                              void* d_out, int out_size, void* d_ws, size_t ws_size,
                              hipStream_t stream) {
  (void)in_sizes; (void)n_in; (void)out_size; (void)ws_size;
  const float* x     = (const float*)d_in[0];
  const float* ln_w  = (const float*)d_in[3];
  const float* ln_b  = (const float*)d_in[4];
  const float* w_qkv = (const float*)d_in[5];
  const float* w_out = (const float*)d_in[6];
  const float* b_out = (const float*)d_in[7];
  float* y = (float*)d_out;
  bf16* ws = (bf16*)d_ws;

  ln_k<<<512, 256, 0, stream>>>(x, ln_w, ln_b, ws + XN_OFF);
  qkv_k<<<dim3(128, 24), 256, 0, stream>>>(ws + XN_OFF, w_qkv, ws);
  attn_k<<<dim3(8, 128), 256, 0, stream>>>(ws, ws);
  out_k<<<dim3(128, 4), 256, 0, stream>>>(ws, w_out, b_out, y);
}
